// CrfLossLayer_72395968741563
// MI455X (gfx1250) — matrix operand order repaired and checked by reading, not compiled, not run on hardware
//
#include <hip/hip_runtime.h>
#include <hip/hip_bf16.h>
#include <cstdint>

// CRF loss + Viterbi for B=256, T=512, K=128 on gfx1250.
//
// Forward algorithm recast as per-timestep f16 GEMM via the identity
//   logsumexp_i(alpha[i]+T[i,j]) = log( sum_i exp(alpha[i]-m) * exp(T[i,j]) ) + m
// so each of the 511 sequential steps is a (256x128)x(128x128) GEMM done with
// v_wmma_f32_16x16x32_f16. The next step's emission tile is pulled into LDS
// with global_load_async_to_lds_b128 (ASYNCcnt) while the current step's
// WMMAs run.
//
// Workspace layout (bytes):
//   [0      , 32768 )  E_h   : f16 exp(transitions), row-major 128x128
//   [32768  , 65536 )  T_h   : f16 transitions (for Viterbi LDS tile)
//   [65536  , 66560 )  logZ  : f32[256]
//   [66560  , 67584 )  score : f32[256]
//   [67584  , 68608 )  lastTag : i32[256]
//   [68608  , 68612 )  correct : u32 counter (zeroed each launch by prep)
//   [131072 , 131072+256*512*128 )  bp : u8 backpointers
//
// d_out (float): [0]=loss, [1 .. 1+B*T)=decoded tags, [1+B*T]=tag_accuracy.

#define BDIM 256
#define TDIM 512
#define KDIM 128
#define ASTR 136  // padded f16 row stride for the WMMA A staging tile

typedef __attribute__((ext_vector_type(16))) _Float16 v16h;
typedef __attribute__((ext_vector_type(8)))  float    v8f;

union AFrag { v16h v; unsigned u[8]; };
union BFrag { v16h v; uint4    q[2]; };

// Async global->LDS 16-byte copy (ASYNCcnt path). Generic pointers to LDS on
// AMDGPU carry the LDS byte offset in their low 32 bits, which is exactly the
// VDST operand of global_load_async_to_lds_*.
__device__ __forceinline__ void async_ld_f4(void* ldsDst, const float* gsrc) {
#if defined(__gfx1250__)
    unsigned lds_off = (unsigned)(uintptr_t)ldsDst;
    unsigned long long ga = (unsigned long long)(uintptr_t)gsrc;
    asm volatile("global_load_async_to_lds_b128 %0, %1, off"
                 :: "v"(lds_off), "v"(ga) : "memory");
#else
    *(float4*)ldsDst = *(const float4*)gsrc;
#endif
}
__device__ __forceinline__ void async_wait0() {
#if defined(__gfx1250__)
    asm volatile("s_wait_asynccnt 0x0" ::: "memory");
#endif
}

// ---------------------------------------------------------------------------
// Prep: E = exp(trans) in f16, T in f16, zero the accuracy counter.
// ---------------------------------------------------------------------------
__global__ void crf_prep_kernel(const float* __restrict__ trans,
                                _Float16* __restrict__ E_h,
                                _Float16* __restrict__ T_h,
                                unsigned int* __restrict__ correct) {
    int i = blockIdx.x * blockDim.x + threadIdx.x;
    if (i == 0) *correct = 0u;
    if (i < KDIM * KDIM) {
        float x = trans[i];
        E_h[i] = (_Float16)__expf(x);
        T_h[i] = (_Float16)x;
    }
}

// ---------------------------------------------------------------------------
// Gold-path sequence score: unary + binary sums under the length mask.
// ---------------------------------------------------------------------------
__global__ void crf_score_kernel(const float* __restrict__ emis,
                                 const int* __restrict__ tags,
                                 const int* __restrict__ lengths,
                                 const float* __restrict__ trans,
                                 float* __restrict__ score) {
    int b = blockIdx.x * blockDim.x + threadIdx.x;
    if (b >= BDIM) return;
    int len = lengths[b];
    int prev = tags[b * TDIM];
    float s = emis[(size_t)b * TDIM * KDIM + prev];  // len >= 1 always
    for (int t = 1; t < TDIM; ++t) {
        int tg = tags[b * TDIM + t];
        if (t < len)
            s += emis[((size_t)b * TDIM + t) * KDIM + tg] + trans[prev * KDIM + tg];
        prev = tg;
    }
    score[b] = s;
}

// ---------------------------------------------------------------------------
// Forward (log-partition) via per-step WMMA GEMM.
// One block = 16 batch rows; 8 waves, wave w owns output columns [16w,16w+16).
// Emission tile for step t+1 is async-copied to LDS while step t computes.
// ---------------------------------------------------------------------------
__global__ __launch_bounds__(256) void crf_forward_kernel(
        const float* __restrict__ emis,
        const int* __restrict__ lengths,
        const _Float16* __restrict__ E_h,
        float* __restrict__ logZ) {
    __shared__ float    alpha[16][KDIM];
    __shared__ _Float16 af16[16][ASTR];
    __shared__ float    emisLds[2][16][KDIM];  // double-buffered emission tile
    __shared__ float    red[16][16];
    __shared__ float    rowmax[16];
    __shared__ int      slen[16];

    const int tid  = threadIdx.x;
    const int wave = tid >> 5;
    const int lane = tid & 31;
    const int b0   = blockIdx.x * 16;
    const int n0   = wave * 16;
    const int h    = lane >> 4;   // which half of the wave
    const int l15  = lane & 15;
    const int r    = tid >> 4;    // staging/reduction row (0..15)
    const int c16  = tid & 15;    // reduction column slot
    const int ccol = c16 * 8;     // this thread's 8-float slice of a tile row

    // Constant B fragments: E tile rows [32ks, 32ks+32), cols [n0, n0+16).
    // Lane L holds column n0+L%16; element s is row k0+(s>>3)*16+8*(L/16)+(s&7).
    BFrag bf[4];
#pragma unroll
    for (int ks = 0; ks < 4; ++ks) {
#pragma unroll
        for (int s = 0; s < 16; ++s) {
            int k = ks * 32 + (s >> 3) * 16 + h * 8 + (s & 7);
            bf[ks].v[s] = E_h[k * KDIM + n0 + l15];
        }
    }

    // alpha <- emissions[:,0]; kick off async copy of the t=1 emission tile.
    const float* emisRow = emis + (size_t)(b0 + r) * TDIM * KDIM;  // row r base
#pragma unroll
    for (int i = 0; i < 8; ++i) {
        int c = c16 + 16 * i;
        alpha[r][c] = emisRow[c];
    }
    async_ld_f4(&emisLds[1][r][ccol],     emisRow + (size_t)1 * KDIM + ccol);
    async_ld_f4(&emisLds[1][r][ccol + 4], emisRow + (size_t)1 * KDIM + ccol + 4);
    if (tid < 16) slen[tid] = lengths[b0 + tid];
    async_wait0();
    __syncthreads();

    // Per-lane copies of the lengths for the 8 output rows this lane owns.
    int mylen[8];
#pragma unroll
    for (int rr = 0; rr < 8; ++rr) mylen[rr] = slen[rr + 8 * h];

    for (int t = 1; t < TDIM; ++t) {
        // --- prefetch next emission tile (overlaps everything below) ---
        if (t + 1 < TDIM) {
            const float* g = emisRow + (size_t)(t + 1) * KDIM + ccol;
            float* d = &emisLds[(t + 1) & 1][r][ccol];
            async_ld_f4(d, g);
            async_ld_f4(d + 4, g + 4);
        }

        // --- row maxima ---
        float m = alpha[r][c16];
#pragma unroll
        for (int i = 1; i < 8; ++i) m = fmaxf(m, alpha[r][c16 + 16 * i]);
        red[r][c16] = m;
        __syncthreads();
        if (tid < 16) {
            float mm = red[tid][0];
#pragma unroll
            for (int i = 1; i < 16; ++i) mm = fmaxf(mm, red[tid][i]);
            rowmax[tid] = mm;
        }
        __syncthreads();

        // --- a = exp(alpha - rowmax) in f16 (WMMA A staging) ---
        {
            float rm = rowmax[r];
#pragma unroll
            for (int i = 0; i < 8; ++i) {
                int c = c16 + 16 * i;
                af16[r][c] = (_Float16)__expf(alpha[r][c] - rm);
            }
        }
        __syncthreads();

        // --- 16x16 tile of (a x exp(T)) via 4 chained WMMAs over K=128 ---
        v8f acc = {};
#pragma unroll
        for (int ks = 0; ks < 4; ++ks) {
            AFrag af;
            int kbase = ks * 32 + h * 8;
#pragma unroll
            for (int i = 0; i < 8; ++i) {
                int K = kbase + (i >> 2) * 16 + (i & 3) * 2;  // even -> dword aligned
                af.u[i] = *(const unsigned*)&af16[l15][K];
            }
            acc = __builtin_amdgcn_wmma_f32_16x16x32_f16(
                false, af.v, false, bf[ks].v, (short)0, acc, false, false);
        }

        // --- branchless masked update: alpha = valid ? log+max+emit : alpha ---
#pragma unroll
        for (int rr = 0; rr < 8; ++rr) {
            int mrow = rr + 8 * h;
            float oldv = alpha[mrow][n0 + l15];
            float newv = __logf(acc[rr]) + rowmax[mrow] +
                         emisLds[t & 1][mrow][n0 + l15];
            alpha[mrow][n0 + l15] = (t < mylen[rr]) ? newv : oldv;
        }

        async_wait0();     // next tile resident before anyone reads it
        __syncthreads();
    }

    // --- final logsumexp over states ---
    float m = alpha[r][c16];
#pragma unroll
    for (int i = 1; i < 8; ++i) m = fmaxf(m, alpha[r][c16 + 16 * i]);
    red[r][c16] = m;
    __syncthreads();
    if (tid < 16) {
        float mm = red[tid][0];
#pragma unroll
        for (int i = 1; i < 16; ++i) mm = fmaxf(mm, red[tid][i]);
        rowmax[tid] = mm;
    }
    __syncthreads();
    float s = 0.f;
    {
        float rm = rowmax[r];
#pragma unroll
        for (int i = 0; i < 8; ++i) s += __expf(alpha[r][c16 + 16 * i] - rm);
    }
    red[r][c16] = s;
    __syncthreads();
    if (tid < 16) {
        float ss = 0.f;
#pragma unroll
        for (int i = 0; i < 16; ++i) ss += red[tid][i];
        logZ[b0 + tid] = rowmax[tid] + __logf(ss);
    }
}

// ---------------------------------------------------------------------------
// Viterbi forward (max-plus; no tensor-core path). 2 batches per block,
// trans cached in LDS as f16 (32 KB). Thread j owns next-state j.
// ---------------------------------------------------------------------------
__global__ __launch_bounds__(256) void crf_viterbi_kernel(
        const float* __restrict__ emis,
        const int* __restrict__ lengths,
        const _Float16* __restrict__ T_h,
        unsigned char* __restrict__ bp,
        int* __restrict__ lastTag) {
    __shared__ _Float16 tr[KDIM * KDIM];
    __shared__ float    v[2][2][KDIM];  // [buffer][sub-batch][state]

    const int tid = threadIdx.x;
    const int sub = tid >> 7;
    const int j   = tid & 127;
    const int b   = blockIdx.x * 2 + sub;

    {   // cooperative copy of the f16 transition matrix into LDS
        const unsigned* src = (const unsigned*)T_h;
        unsigned* dst = (unsigned*)tr;
        for (int i = tid; i < KDIM * KDIM / 2; i += 256) dst[i] = src[i];
    }
    const int len = lengths[b];
    v[0][sub][j] = emis[(size_t)b * TDIM * KDIM + j];
    __syncthreads();

    int cur = 0;
    for (int t = 1; t < TDIM; ++t) {
        float vj;
        int   bj;
        if (t < len) {
            const float* vc = v[cur][sub];
            float best = -3.4e38f;
            int   bi   = 0;
#pragma unroll 4
            for (int i = 0; i < KDIM; ++i) {
                float cand = vc[i] + (float)tr[i * KDIM + j];
                if (cand > best) { best = cand; bi = i; }
            }
            vj = best + emis[((size_t)b * TDIM + t) * KDIM + j];
            bj = bi;
        } else {
            vj = v[cur][sub][j];
            bj = j;  // identity backpointer for padded steps
        }
        bp[((size_t)b * TDIM + t) * KDIM + j] = (unsigned char)bj;
        v[cur ^ 1][sub][j] = vj;
        __syncthreads();
        cur ^= 1;
    }

    if (j == 0) {  // argmax of final scores (first-max tie-break, as argmax)
        float best = v[cur][sub][0];
        int   bi   = 0;
        for (int i = 1; i < KDIM; ++i) {
            float x = v[cur][sub][i];
            if (x > best) { best = x; bi = i; }
        }
        lastTag[b] = bi;
    }
}

// ---------------------------------------------------------------------------
// Backtrack + write decoded (as float) + count correct tags.
// ---------------------------------------------------------------------------
__global__ void crf_backtrack_kernel(const int* __restrict__ tags,
                                     const int* __restrict__ lengths,
                                     const unsigned char* __restrict__ bp,
                                     const int* __restrict__ lastTag,
                                     float* __restrict__ out_decoded,
                                     unsigned int* __restrict__ correct) {
    int b = threadIdx.x;
    int len = lengths[b];
    int tag = lastTag[b];
    unsigned int c = 0;
    for (int t = TDIM - 1; t >= 1; --t) {
        out_decoded[(size_t)b * TDIM + t] = (float)tag;
        if (t < len && tags[b * TDIM + t] == tag) ++c;
        tag = bp[((size_t)b * TDIM + t) * KDIM + tag];
    }
    out_decoded[(size_t)b * TDIM] = (float)tag;
    if (tags[b * TDIM] == tag) ++c;  // t=0 always valid (len >= 1)
    atomicAdd(correct, c);
}

// ---------------------------------------------------------------------------
// Finalize: loss = mean(logZ - score); accuracy = correct / sum(len).
// ---------------------------------------------------------------------------
__global__ void crf_final_kernel(const float* __restrict__ logZ,
                                 const float* __restrict__ score,
                                 const int* __restrict__ lengths,
                                 const unsigned int* __restrict__ correct,
                                 float* __restrict__ out_loss,
                                 float* __restrict__ out_acc) {
    __shared__ float rs[BDIM];
    __shared__ int   rl[BDIM];
    int b = threadIdx.x;
    rs[b] = logZ[b] - score[b];
    rl[b] = lengths[b];
    __syncthreads();
    for (int s = BDIM / 2; s > 0; s >>= 1) {
        if (b < s) { rs[b] += rs[b + s]; rl[b] += rl[b + s]; }
        __syncthreads();
    }
    if (b == 0) {
        *out_loss = rs[0] / (float)BDIM;
        *out_acc  = (float)(*correct) / (float)rl[0];
    }
}

// ---------------------------------------------------------------------------
extern "C" void kernel_launch(void* const* d_in, const int* in_sizes, int n_in,
                              void* d_out, int out_size, void* d_ws, size_t ws_size,
                              hipStream_t stream) {
    const float* emis  = (const float*)d_in[0];  // (B,T,K) f32
    const int*   tags  = (const int*)d_in[1];    // (B,T)   i32
    const int*   lens  = (const int*)d_in[2];    // (B,)    i32
    const float* trans = (const float*)d_in[3];  // (K,K)   f32
    float* out = (float*)d_out;

    char* ws = (char*)d_ws;
    _Float16*      E_h     = (_Float16*)(ws + 0);
    _Float16*      T_h     = (_Float16*)(ws + 32768);
    float*         logZ    = (float*)(ws + 65536);
    float*         score   = (float*)(ws + 66560);
    int*           lastTag = (int*)(ws + 67584);
    unsigned int*  correct = (unsigned int*)(ws + 68608);
    unsigned char* bp      = (unsigned char*)(ws + 131072);

    crf_prep_kernel<<<(KDIM * KDIM + 255) / 256, 256, 0, stream>>>(trans, E_h, T_h, correct);
    crf_score_kernel<<<1, BDIM, 0, stream>>>(emis, tags, lens, trans, score);
    crf_forward_kernel<<<BDIM / 16, 256, 0, stream>>>(emis, lens, E_h, logZ);
    crf_viterbi_kernel<<<BDIM / 2, 256, 0, stream>>>(emis, lens, T_h, bp, lastTag);
    crf_backtrack_kernel<<<1, BDIM, 0, stream>>>(tags, lens, bp, lastTag, out + 1, correct);
    crf_final_kernel<<<1, BDIM, 0, stream>>>(logZ, score, lens, correct,
                                             out, out + (out_size - 1));
}
